// PoolHiddenNet_223338299636
// MI455X (gfx1250) — matrix operand order repaired and checked by reading, not compiled, not run on hardware
//
#include <hip/hip_runtime.h>
#include <hip/hip_bf16.h>

typedef _Float16 v16h __attribute__((ext_vector_type(16)));
typedef float    v8f  __attribute__((ext_vector_type(8)));
typedef unsigned int u32x4 __attribute__((ext_vector_type(4)));
typedef int          i32x8 __attribute__((ext_vector_type(8)));
typedef int          i32x4 __attribute__((ext_vector_type(4)));

#define S_    16
#define P_    32
#define H_    64
#define E_    64
#define KIN   128        // E + H
#define D1_   8192
#define BOT_  1024
#define ROWS_ 16384      // S*P*P
#define LDSTR 136        // padded LDS row stride in halfwords (8*17 -> keeps 16B align, spreads banks)
#define W1SL  (128 * LDSTR)   // halfwords per W1 slice buffer

// ---------------- prep: x = [ emb(clip(pos_b - pos_a)) | h_b ] in f16 ----------------
__global__ void prep_x_kernel(const float* __restrict__ h,
                              const float* __restrict__ pos,
                              const float* __restrict__ Wsp,
                              const float* __restrict__ bsp,
                              _Float16* __restrict__ X)
{
    int idx = blockIdx.x * 256 + threadIdx.x;   // ROWS_*KIN elements total
    int row = idx >> 7;
    int k   = idx & (KIN - 1);
    int s = row >> 10;
    int a = (row >> 5) & 31;
    int b = row & 31;
    int pa = s * P_ + a;
    int pb = s * P_ + b;
    float v;
    if (k < E_) {
        // NB=2 -> half=1: clip to [-1,1], divide by 1
        float rx = pos[2 * pb]     - pos[2 * pa];
        float ry = pos[2 * pb + 1] - pos[2 * pa + 1];
        rx = fminf(fmaxf(rx, -1.f), 1.f);
        ry = fminf(fmaxf(ry, -1.f), 1.f);
        v = rx * Wsp[k] + ry * Wsp[E_ + k] + bsp[k];
    } else {
        v = h[pb * H_ + (k - E_)];
    }
    X[idx] = (_Float16)v;
}

// ---------------- prep: fp32 row-major (K x N) -> f16 "col-major" [N][K] ----------------
__global__ void transpose_cvt_kernel(const float* __restrict__ in,
                                     _Float16* __restrict__ out,
                                     int K, int N)
{
    __shared__ float tile[32][33];
    int n0 = blockIdx.x * 32;
    int k0 = blockIdx.y * 32;
#pragma unroll
    for (int i = 0; i < 4; ++i)
        tile[threadIdx.y + 8 * i][threadIdx.x] =
            in[(size_t)(k0 + threadIdx.y + 8 * i) * N + n0 + threadIdx.x];
    __syncthreads();
#pragma unroll
    for (int i = 0; i < 4; ++i)
        out[(size_t)(n0 + threadIdx.y + 8 * i) * K + k0 + threadIdx.x] =
            (_Float16)tile[threadIdx.x][threadIdx.y + 8 * i];
}

// ---------------- WMMA fragment loaders (contiguous k order, same for A and B, wave32) ----------------
// A (16x32 f16): lane L -> row M = L&15; 16 contiguous K starting at (L>>4)*16
__device__ __forceinline__ v16h load_a_frag(const _Float16* base, int row, int kbase, int lh)
{
    const _Float16* p = base + row * LDSTR + kbase + lh * 16;
    v16h a;
#pragma unroll
    for (int i = 0; i < 8; ++i) { a[i] = p[i]; a[8 + i] = p[8 + i]; }
    return a;
}
// B (32x16 f16): lane L -> col N = L&15; 16 contiguous K starting at (L>>4)*16
__device__ __forceinline__ v16h load_b_frag16(const _Float16* p)
{
    v16h b;
#pragma unroll
    for (int i = 0; i < 16; ++i) b[i] = p[i];
    return b;
}

// ---------------- TDM: async DMA of one 128x128 f16 W1T slice global -> LDS ----------------
// D# per ISA 8.3/8.4: 2D tensor, data_size=2B, tensor_dim0=128 (stride 128), tile 128x128,
// LDS padding 4 DWORDs every 64 DWORDs -> reproduces the LDSTR=136 row stride.
__device__ __forceinline__ void tdm_load_w1_slice(const _Float16* gsrc, _Float16* ldst)
{
    unsigned long long ga = (unsigned long long)(uintptr_t)gsrc;
    unsigned int       la = (unsigned int)(uintptr_t)ldst;   // low 32 bits = LDS byte offset
    u32x4 g0;
    g0[0] = 1u;                                              // count=1 (valid user D#)
    g0[1] = la;                                              // lds_addr
    g0[2] = (unsigned int)ga;                                // global_addr[31:0]
    g0[3] = (unsigned int)((ga >> 32) & 0x1FFFFFFu)          // global_addr[56:32]
          | (2u << 30);                                      // type = 2 ("image")
    i32x8 g1;
    g1[0] = (1 << 16)                                        // data_size = 1 -> 2 bytes
          | (1 << 20)                                        // pad_enable
          | (5 << 22)                                        // pad_interval: 64 DWORDs (256B row)
          | (3 << 25);                                       // pad_amount:   4 DWORDs (16B)
    g1[1] = (int)(128u << 16);                               // tensor_dim0[15:0]=128
    g1[2] = (int)(8192u << 16);                              // dim0 hi=0 | tensor_dim1[15:0]=8192
    g1[3] = (int)(128u << 16);                               // dim1 hi=0 | tile_dim0=128
    g1[4] = (int)(128u);                                     // tile_dim1=128 | tile_dim2=0
    g1[5] = (int)(128u);                                     // tensor_dim0_stride = 128
    g1[6] = 0;
    g1[7] = 0;
    i32x4 z4 = {0, 0, 0, 0};                                 // groups 2/3 unused (2D tensor)
    i32x8 z8 = {0, 0, 0, 0, 0, 0, 0, 0};
    __builtin_amdgcn_tensor_load_to_lds(g0, g1, z4, z4, z8, 0);
}

// ---------------- fused layer1 -> layer2 -> max-pool kernel ----------------
__launch_bounds__(512, 1)
__global__ void pool_fused_kernel(const _Float16* __restrict__ X,
                                  const _Float16* __restrict__ W1T,
                                  const _Float16* __restrict__ W2T,
                                  const float* __restrict__ b1,
                                  const float* __restrict__ b2,
                                  float* __restrict__ out)
{
    __shared__ _Float16 lds_x[64 * LDSTR];   // 64 x 128 x-tile (row-major, padded)
    __shared__ _Float16 w1s [2 * W1SL];      // double-buffered W1T slice: 128 x 128 each
    __shared__ _Float16 y1s [64 * LDSTR];    // relu(x@W1) chunk, 64 x 128

    const int tid  = threadIdx.x;
    const int wave = tid >> 5;               // 0..15, owns output cols [wave*64, wave*64+64)
    const int lane = tid & 31;
    const int lr   = lane & 15;
    const int lh   = lane >> 4;
    const int c    = blockIdx.x;             // row chunk: 64 rows = 2 'a'-groups of one scene
    const int row0 = c * 64;

    // kick off the async DMA of the first W1T slice (wave-level op, wave 0 only)
    if (wave == 0)
        tdm_load_w1_slice(W1T, w1s);

    // stage the 64x128 x-tile (contiguous 16KB region, 32B per thread) - overlaps the DMA
    {
        const _Float16* src = X + (size_t)row0 * KIN + tid * 16;
        _Float16* dst = lds_x + ((tid * 16) >> 7) * LDSTR + ((tid * 16) & (KIN - 1));
#pragma unroll
        for (int i = 0; i < 16; ++i) dst[i] = src[i];
    }

    v8f acc[4][4];
#pragma unroll
    for (int tm = 0; tm < 4; ++tm)
#pragma unroll
        for (int tn = 0; tn < 4; ++tn)
            acc[tm][tn] = {};

    for (int it = 0; it < D1_ / 128; ++it) {
        const int kc  = it * 128;
        const int buf = it & 1;
        const _Float16* w1cur = w1s + buf * W1SL;

        if (wave == 0)
            __builtin_amdgcn_s_wait_tensorcnt(0);   // current slice landed in LDS
        __syncthreads();   // slice visible to all; previous iter's y1s readers done

        // prefetch next slice into the other buffer while we compute on this one
        if (wave == 0 && it + 1 < D1_ / 128)
            tdm_load_w1_slice(W1T + (size_t)(kc + 128) * KIN, w1s + (buf ^ 1) * W1SL);

        // layer 1: y1 = relu(x @ W1[:,kc:kc+128] + b1), 64x128, 2 tiles per wave
#pragma unroll
        for (int tt = 0; tt < 2; ++tt) {
            const int t  = wave * 2 + tt;
            const int tm = t & 3;            // 0..3 over 64 rows
            const int tn = t >> 2;           // 0..7 over 128 chunk cols
            v8f y = {};
#pragma unroll
            for (int ks = 0; ks < 4; ++ks) {
                v16h a = load_a_frag(lds_x, tm * 16 + lr, ks * 32, lh);
                v16h b = load_b_frag16(w1cur + (tn * 16 + lr) * LDSTR + ks * 32 + lh * 16);
                y = __builtin_amdgcn_wmma_f32_16x16x32_f16(false, a, false, b,
                                                           (short)0, y, false, false);
            }
            const float bias = b1[kc + tn * 16 + lr];
#pragma unroll
            for (int r = 0; r < 8; ++r) {
                float v = y[r] + bias;
                v = v > 0.f ? v : 0.f;
                // C/D layout: VGPR r -> row r + 8*lh within tile, col = lr
                y1s[(tm * 16 + lh * 8 + r) * LDSTR + tn * 16 + lr] = (_Float16)v;
            }
        }
        __syncthreads();   // y1s ready

        // layer 2 partial: acc += y1(64x128) @ W2T[kc:kc+128, wave's 64 cols]
#pragma unroll
        for (int ks = 0; ks < 4; ++ks) {
            v16h a[4];
#pragma unroll
            for (int tm = 0; tm < 4; ++tm)
                a[tm] = load_a_frag(y1s, tm * 16 + lr, ks * 32, lh);
            v16h b[4];
#pragma unroll
            for (int tn = 0; tn < 4; ++tn) {
                const _Float16* p = W2T + (size_t)(wave * 64 + tn * 16 + lr) * D1_
                                        + kc + ks * 32 + lh * 16;   // L2-resident stream
                b[tn] = load_b_frag16(p);
            }
#pragma unroll
            for (int tm = 0; tm < 4; ++tm)
#pragma unroll
                for (int tn = 0; tn < 4; ++tn)
                    acc[tm][tn] = __builtin_amdgcn_wmma_f32_16x16x32_f16(
                        false, a[tm], false, b[tn], (short)0, acc[tm][tn], false, false);
        }
    }

    // epilogue: max over the 32 'b' rows of each of the 2 'a'-groups, +b2, relu, store
#pragma unroll
    for (int g = 0; g < 2; ++g) {
#pragma unroll
        for (int tn = 0; tn < 4; ++tn) {
            const int n = wave * 64 + tn * 16 + lr;
            float m = -3.4e38f;
#pragma unroll
            for (int t2 = 0; t2 < 2; ++t2) {
                const int tm = g * 2 + t2;
#pragma unroll
                for (int r = 0; r < 8; ++r) m = fmaxf(m, acc[tm][tn][r]);
            }
            m += b2[n];                       // max(x)+b == max(x+b), bias uniform per column
            m = m > 0.f ? m : 0.f;            // relu after max == max after relu (monotone)
            m = fmaxf(m, __shfl_xor(m, 16, 32));   // merge lane halves (rows +8/+24)
            if (lh == 0)
                out[(size_t)(c * 2 + g) * BOT_ + n] = m;
        }
    }
}

extern "C" void kernel_launch(void* const* d_in, const int* in_sizes, int n_in,
                              void* d_out, int out_size, void* d_ws, size_t ws_size,
                              hipStream_t stream) {
    const float* h   = (const float*)d_in[0];   // (1,B,H)
    const float* pos = (const float*)d_in[1];   // (B,2) end_pos
    const float* Wsp = (const float*)d_in[4];   // (2,E)
    const float* bsp = (const float*)d_in[5];   // (E)
    const float* W1  = (const float*)d_in[6];   // (128, 8192) row-major
    const float* b1  = (const float*)d_in[7];   // (8192)
    const float* W2  = (const float*)d_in[8];   // (8192, 1024) row-major
    const float* b2  = (const float*)d_in[9];   // (1024)
    float* out = (float*)d_out;

    char* ws = (char*)d_ws;
    _Float16* X   = (_Float16*)(ws);                      //  4 MiB: 16384 x 128 f16
    _Float16* W1T = (_Float16*)(ws + (size_t)(4 << 20));  //  2 MiB: [8192][128] f16
    _Float16* W2T = (_Float16*)(ws + (size_t)(6 << 20));  // 16 MiB: [1024][8192] f16

    prep_x_kernel<<<(ROWS_ * KIN) / 256, 256, 0, stream>>>(h, pos, Wsp, bsp, X);
    transpose_cvt_kernel<<<dim3(D1_ / 32, KIN / 32), dim3(32, 8), 0, stream>>>(W1, W1T, KIN, D1_);
    transpose_cvt_kernel<<<dim3(BOT_ / 32, D1_ / 32), dim3(32, 8), 0, stream>>>(W2, W2T, D1_, BOT_);
    pool_fused_kernel<<<ROWS_ / 64, 512, 0, stream>>>(X, W1T, W2T, b1, b2, out);
}
